// CausalAttention_51934744543525
// MI455X (gfx1250) — matrix operand order repaired and checked by reading, not compiled, not run on hardware
//
#include <hip/hip_runtime.h>

typedef __attribute__((ext_vector_type(16))) _Float16 v16h;
typedef __attribute__((ext_vector_type(8)))  _Float16 v8h;
typedef __attribute__((ext_vector_type(4)))  _Float16 v4h;
typedef __attribute__((ext_vector_type(8)))  float    v8f;
typedef __attribute__((ext_vector_type(4)))  float    v4f;

#define DDIM   1024
#define NSEQ   2048
#define NBATCH 8

// ---------------------------------------------------------------------------
// WMMA fragment loaders (CDNA5 16-bit layouts, wave32).
// A (16x32, MxK), source row-major with leading dim `ld` (halfs):
//   lanes 0-15  : row = lane,    halfs[0..7]=K 0..7,   halfs[8..15]=K 16..23
//   lanes 16-31 : row = lane-16, halfs[0..7]=K 8..15,  halfs[8..15]=K 24..31
// B (32x16, KxN), source stored N-major (each "row" = one output column, K contiguous):
//   lanes 0-15  : col = lane,    halfs[0..7]=K 0..7,   halfs[8..15]=K 16..23
//   lanes 16-31 : col = lane-16, halfs[0..7]=K 8..15,  halfs[8..15]=K 24..31
// ---------------------------------------------------------------------------
__device__ __forceinline__ v16h load_frag_a(const _Float16* base, int ld) {
  const int lane = threadIdx.x & 31;
  const int lo = lane & 15, hi = (lane >> 4) & 1;
  const _Float16* p = base + (size_t)lo * ld + hi * 8;
  v8h a0 = *(const v8h*)(p);
  v8h a1 = *(const v8h*)(p + 16);
  v16h f;
#pragma unroll
  for (int i = 0; i < 8; ++i) { f[i] = a0[i]; f[i + 8] = a1[i]; }
  return f;
}

__device__ __forceinline__ v16h load_frag_b(const _Float16* base, int ld) {
  const int lane = threadIdx.x & 31;
  const int lo = lane & 15, hi = (lane >> 4) & 1;
  const _Float16* p = base + (size_t)lo * ld + hi * 8;
  v8h b0 = *(const v8h*)(p);
  v8h b1 = *(const v8h*)(p + 16);
  v16h f;
#pragma unroll
  for (int i = 0; i < 8; ++i) { f[i] = b0[i]; f[i + 8] = b1[i]; }
  return f;
}

__device__ __forceinline__ v8f wmma_f16(v16h a, v16h b, v8f c) {
  return __builtin_amdgcn_wmma_f32_16x16x32_f16(false, a, false, b, (short)0, c,
                                                false, false);
}

// ---------------------------------------------------------------------------
// Stage 0: casts
// ---------------------------------------------------------------------------
__global__ void cast_f32_f16_kernel(const float* __restrict__ in,
                                    _Float16* __restrict__ out, size_t n4) {
  size_t i = (size_t)blockIdx.x * blockDim.x + threadIdx.x;
  if (i >= n4) return;
  v4f v = *(const v4f*)(in + i * 4);
  v4h h;
#pragma unroll
  for (int j = 0; j < 4; ++j) h[j] = (_Float16)v[j];
  *(v4h*)(out + i * 4) = h;
}

// out[c][d] = (f16) in[d][c]   (1024x1024 weight transpose-cast)
__global__ void castT_kernel(const float* __restrict__ in,
                             _Float16* __restrict__ out) {
  int idx  = blockIdx.x * 256 + threadIdx.x;  // 0 .. 1024*1024-1
  int dcol = idx & 1023;                      // contiguous in output
  int crow = idx >> 10;
  out[idx] = (_Float16)in[(size_t)dcol * DDIM + crow];
}

// ---------------------------------------------------------------------------
// Stage 1: fused QKV projection GEMM.
//   [16384 x 1024] (x, f16) @ [1024 x 1024] (W^T-stored, f16) -> f16
// Block: 256 threads = 8 waves (2x4); block tile 64x128, wave tile 32x32.
// z=0 -> Q row-major, z=1 -> K row-major, z=2 -> V stored transposed [b][d][n].
// ---------------------------------------------------------------------------
__global__ __launch_bounds__(256) void qkv_gemm_kernel(
    const _Float16* __restrict__ xh,
    const _Float16* __restrict__ wqT, const _Float16* __restrict__ wkT,
    const _Float16* __restrict__ wvT,
    _Float16* __restrict__ Qh, _Float16* __restrict__ Kh,
    _Float16* __restrict__ Vt) {
  const int w    = threadIdx.x >> 5;
  const int wm   = w >> 2;                 // 0..1
  const int wn   = w & 3;                  // 0..3
  const int lane = threadIdx.x & 31;
  const int lo = lane & 15, hi = (lane >> 4) & 1;

  const int m0  = blockIdx.y * 64 + wm * 32;
  const int n0  = blockIdx.x * 128 + wn * 32;
  const int sel = blockIdx.z;
  const _Float16* wT = (sel == 0) ? wqT : (sel == 1) ? wkT : wvT;

  v8f acc[2][2];
#pragma unroll
  for (int r = 0; r < 2; ++r)
#pragma unroll
    for (int c = 0; c < 2; ++c)
#pragma unroll
      for (int i = 0; i < 8; ++i) acc[r][c][i] = 0.0f;

  for (int k0 = 0; k0 < DDIM; k0 += 32) {
    v16h a[2], b[2];
#pragma unroll
    for (int r = 0; r < 2; ++r)
      a[r] = load_frag_a(xh + (size_t)(m0 + 16 * r) * DDIM + k0, DDIM);
#pragma unroll
    for (int c = 0; c < 2; ++c)
      b[c] = load_frag_b(wT + (size_t)(n0 + 16 * c) * DDIM + k0, DDIM);
#pragma unroll
    for (int r = 0; r < 2; ++r)
#pragma unroll
      for (int c = 0; c < 2; ++c) acc[r][c] = wmma_f16(a[r], b[c], acc[r][c]);
  }

  // ---- epilogue: selection hoisted out of the store loops (straight-line
  //      store clauses instead of per-element branches)
  if (sel < 2) {
    _Float16* __restrict__ outp = (sel == 0) ? Qh : Kh;
#pragma unroll
    for (int r = 0; r < 2; ++r) {
      const int row = m0 + 16 * r + hi * 8;
#pragma unroll
      for (int c = 0; c < 2; ++c) {
        const int col = n0 + 16 * c + lo;
#pragma unroll
        for (int i = 0; i < 8; ++i)
          outp[(size_t)(row + i) * DDIM + col] = (_Float16)acc[r][c][i];
      }
    }
  } else {
    // V^T store: one lane's 8 accumulator rows are contiguous along n ->
    // pack into a single 16-byte store.
#pragma unroll
    for (int r = 0; r < 2; ++r) {
      const int row = m0 + 16 * r + hi * 8;   // multiple of 8; stays in batch
      const int bb  = row >> 11;
      const int nn  = row & 2047;
#pragma unroll
      for (int c = 0; c < 2; ++c) {
        const int col = n0 + 16 * c + lo;
        _Float16* vp = Vt + ((size_t)bb * DDIM + col) * NSEQ + nn;
        v8h hv;
#pragma unroll
        for (int i = 0; i < 8; ++i) hv[i] = (_Float16)acc[r][c][i];
        *(v8h*)vp = hv;
      }
    }
  }
}

// ---------------------------------------------------------------------------
// Stage 2: flash attention. Block = (batch, 32 q-rows), 256 threads = 8 waves.
// Phase 1: 32x64 S-tile as 8 independent 16x16 WMMA tiles (wave -> (qi,kj)),
//          fragments streamed directly from global Q/K (L2-resident).
// Phase 2: online softmax in LDS; each wave owns a 128-wide d-slice of O
//          (2x8 f32 accum tiles), P from LDS (A-frags), V^T from global (B-frags).
// ---------------------------------------------------------------------------
__global__ __launch_bounds__(256) void flash_attn_kernel(
    const _Float16* __restrict__ Qh, const _Float16* __restrict__ Kh,
    const _Float16* __restrict__ Vt, float* __restrict__ out) {
  __shared__ alignas(16) float    sS[32][64];
  __shared__ alignas(16) _Float16 sP[32][64];
  __shared__ float sm[32], sl[32], sscale[32];

  const int w    = threadIdx.x >> 5;
  const int lane = threadIdx.x & 31;
  const int lo = lane & 15, hi = (lane >> 4) & 1;
  const int qi = w >> 2, kj = w & 3;   // phase-1 S-tile coords
  const int c0 = w * 128;              // phase-2 d-slice base

  const int b  = blockIdx.y;
  const int q0 = blockIdx.x * 32;

  const _Float16* Qb = Qh + (size_t)b * NSEQ * DDIM;
  const _Float16* Kb = Kh + (size_t)b * NSEQ * DDIM;
  const _Float16* Vb = Vt + (size_t)b * DDIM * NSEQ;

  if (threadIdx.x < 32) { sm[threadIdx.x] = -1e30f; sl[threadIdx.x] = 0.0f; }
  __syncthreads();

  v8f oacc[2][8];
#pragma unroll
  for (int rt = 0; rt < 2; ++rt)
#pragma unroll
    for (int ct = 0; ct < 8; ++ct)
#pragma unroll
      for (int i = 0; i < 8; ++i) oacc[rt][ct][i] = 0.0f;

  for (int kv0 = 0; kv0 <= q0 + 31; kv0 += 64) {
    // prefetch next kv tile's K rows / V^T strip while this tile computes
    // (lowers to global_prefetch_b8; speculative, dropped if beyond range)
    if (kv0 + 64 <= q0 + 31) {
      __builtin_prefetch(Kb + (size_t)(kv0 + 64 + kj * 16 + lo) * DDIM, 0, 3);
      __builtin_prefetch(Vb + (size_t)(c0 + lane * 4) * NSEQ + kv0 + 64, 0, 3);
    }

    // ---- S = Q K^T (this wave's 16x16 tile), K-loop over full d=1024
    v8f s;
#pragma unroll
    for (int i = 0; i < 8; ++i) s[i] = 0.0f;
    for (int d0 = 0; d0 < DDIM; d0 += 32) {
      v16h a  = load_frag_a(Qb + (size_t)(q0 + qi * 16) * DDIM + d0, DDIM);
      v16h bk = load_frag_b(Kb + (size_t)(kv0 + kj * 16) * DDIM + d0, DDIM);
      s = wmma_f16(a, bk, s);
    }
    // scale 1/sqrt(1024) + causal mask, spill to LDS
#pragma unroll
    for (int i = 0; i < 8; ++i) {
      int r  = qi * 16 + i + hi * 8;         // local q row 0..31
      int cg = kv0 + kj * 16 + lo;           // global kv col
      float v = s[i] * 0.03125f;
      if (cg > q0 + r) v = -1e30f;
      sS[r][kj * 16 + lo] = v;
    }
    __syncthreads();

    // ---- online softmax, one row per thread (threads 0..31)
    if (threadIdx.x < 32) {
      int r = threadIdx.x;
      float mold = sm[r], mx = mold;
#pragma unroll 8
      for (int j = 0; j < 64; ++j) mx = fmaxf(mx, sS[r][j]);
      float sc = __expf(mold - mx);
      float l  = sl[r] * sc;
#pragma unroll 8
      for (int j = 0; j < 64; ++j) {
        float p = __expf(sS[r][j] - mx);
        l += p;
        sP[r][j] = (_Float16)p;
      }
      sm[r] = mx; sl[r] = l; sscale[r] = sc;
    }
    __syncthreads();

    // ---- rescale O accumulators by exp(m_old - m_new)
    float scl[2][8];
#pragma unroll
    for (int rt = 0; rt < 2; ++rt)
#pragma unroll
      for (int i = 0; i < 8; ++i) scl[rt][i] = sscale[rt * 16 + i + hi * 8];
#pragma unroll
    for (int rt = 0; rt < 2; ++rt)
#pragma unroll
      for (int ct = 0; ct < 8; ++ct)
#pragma unroll
        for (int i = 0; i < 8; ++i) oacc[rt][ct][i] *= scl[rt][i];

    // ---- O += P @ V  (P: 32x64 from LDS, V^T: contiguous-in-kv from global)
#pragma unroll
    for (int kk = 0; kk < 2; ++kk) {
      v16h ap[2];
#pragma unroll
      for (int rt = 0; rt < 2; ++rt)
        ap[rt] = load_frag_a(&sP[rt * 16][kk * 32], 64);
#pragma unroll
      for (int ct = 0; ct < 8; ++ct) {
        v16h bv = load_frag_b(
            Vb + (size_t)(c0 + ct * 16) * NSEQ + kv0 + kk * 32, NSEQ);
#pragma unroll
        for (int rt = 0; rt < 2; ++rt)
          oacc[rt][ct] = wmma_f16(ap[rt], bv, oacc[rt][ct]);
      }
    }
    __syncthreads();
  }

  // ---- epilogue: divide by softmax denominator, write f32 output
  float li[2][8];
#pragma unroll
  for (int rt = 0; rt < 2; ++rt)
#pragma unroll
    for (int i = 0; i < 8; ++i) li[rt][i] = 1.0f / sl[rt * 16 + i + hi * 8];
#pragma unroll
  for (int rt = 0; rt < 2; ++rt)
#pragma unroll
    for (int ct = 0; ct < 8; ++ct)
#pragma unroll
      for (int i = 0; i < 8; ++i) {
        int r   = rt * 16 + i + hi * 8;
        int col = c0 + ct * 16 + lo;
        out[((size_t)b * NSEQ + q0 + r) * DDIM + col] = oacc[rt][ct][i] * li[rt][i];
      }
}

// ---------------------------------------------------------------------------
extern "C" void kernel_launch(void* const* d_in, const int* in_sizes, int n_in,
                              void* d_out, int out_size, void* d_ws,
                              size_t ws_size, hipStream_t stream) {
  (void)in_sizes; (void)n_in; (void)out_size; (void)ws_size;
  const float* x  = (const float*)d_in[0];
  const float* Wq = (const float*)d_in[1];
  const float* Wk = (const float*)d_in[2];
  const float* Wv = (const float*)d_in[3];
  float* out = (float*)d_out;

  const size_t M = (size_t)NBATCH * NSEQ;          // 16384
  _Float16* xh  = (_Float16*)d_ws;                 // M x 1024        (32 MB)
  _Float16* wqT = xh + M * DDIM;                   // 1024 x 1024     (2 MB)
  _Float16* wkT = wqT + (size_t)DDIM * DDIM;
  _Float16* wvT = wkT + (size_t)DDIM * DDIM;
  _Float16* Qh  = wvT + (size_t)DDIM * DDIM;       // M x 1024        (32 MB)
  _Float16* Kh  = Qh + M * DDIM;                   // M x 1024        (32 MB)
  _Float16* Vt  = Kh + M * DDIM;                   // B x 1024 x 2048 (32 MB)

  size_t nx4 = M * DDIM / 4;
  cast_f32_f16_kernel<<<dim3((unsigned)((nx4 + 255) / 256)), dim3(256), 0,
                        stream>>>(x, xh, nx4);
  castT_kernel<<<dim3(4096), dim3(256), 0, stream>>>(Wq, wqT);
  castT_kernel<<<dim3(4096), dim3(256), 0, stream>>>(Wk, wkT);
  castT_kernel<<<dim3(4096), dim3(256), 0, stream>>>(Wv, wvT);

  // 64-row x 128-col block tiles over [16384 x 1024], z selects Q/K/V
  qkv_gemm_kernel<<<dim3(DDIM / 128, (unsigned)(M / 64), 3), dim3(256), 0,
                    stream>>>(xh, wqT, wkT, wvT, Qh, Kh, Vt);

  // one block per (32 q-rows, batch)
  flash_attn_kernel<<<dim3(NSEQ / 32, NBATCH), dim3(256), 0, stream>>>(Qh, Kh,
                                                                       Vt, out);
}
